// CustomGNN_23802708755058
// MI455X (gfx1250) — compile-verified, matrix operand order checked by reading, not run on hardware
//
#include <hip/hip_runtime.h>

#define N_FIN  32
#define HID    256
#define N_FOUT 64

typedef __attribute__((ext_vector_type(16))) _Float16 v16h;
typedef __attribute__((ext_vector_type(8)))  _Float16 v8h;
typedef __attribute__((ext_vector_type(8)))  float    v8f;

#define WMMA(a, b, c) __builtin_amdgcn_wmma_f32_16x16x32_f16( \
    false, (a), false, (b), (short)0, (c), false, false)

// Branchless ELU using the native v_exp_f32 transcendental.
// elu(x) = x>0 ? x : exp(x)-1.  Argument clamped to <=0 so the dead side of
// the select stays finite (exp(min(x,0)) <= 1).
static __device__ __forceinline__ float fast_elu(float x) {
    const float e = __builtin_amdgcn_exp2f(fminf(x, 0.0f) * 1.44269504088896f) - 1.0f;
    return (x > 0.0f) ? x : e;
}

// ---------------------------------------------------------------------------
// Fragment loader for 16-bit A/B operands of v_wmma_f32_16x16x32_f16.
// p points at a row-major tile (16 rows selected by lane%16, `stride` halves
// per row). ISA layout: lanes 0-15 hold K = k0+[0..7] and k0+16+[0..7];
// lanes 16-31 the same shifted by +8. Two aligned 16B LDS loads per lane.
// ---------------------------------------------------------------------------
static __device__ __forceinline__ v16h ld_frag(const _Float16* p, int stride, int k0) {
    const int lane = threadIdx.x & 31;
    const _Float16* q = p + (lane & 15) * stride + k0 + ((lane >> 4) << 3);
    v8h lo = *(const v8h*)q;
    v8h hi = *(const v8h*)(q + 16);
    return __builtin_shufflevector(lo, hi, 0, 1, 2, 3, 4, 5, 6, 7,
                                           8, 9, 10, 11, 12, 13, 14, 15);
}

// Store a 16x16 f32 D tile (bias + ELU) as f16 into a row-major LDS tile.
// D layout: lane l holds col N=l%16; element r holds row M = r + 8*(l/16).
// `bias` points at LDS-staged bias values.
static __device__ __forceinline__ void st_elu_f16(_Float16* base, int stride,
                                                  v8f d, const float* bias, int o0) {
    const int lane = threadIdx.x & 31;
    const int col  = lane & 15;
    const int r0   = (lane >> 4) << 3;
    const float bv = bias[o0 + col];
#pragma unroll
    for (int r = 0; r < 8; ++r) {
        base[(r0 + r) * stride + col] = (_Float16)fast_elu(d[r] + bv);
    }
}

// ---------------------------------------------------------------------------
// Edge scatter: msg[dst] += feat[src] * w, one lane per (edge, channel).
// ---------------------------------------------------------------------------
__global__ __launch_bounds__(256) void gnn_scatter(
    const float* __restrict__ feat, const float* __restrict__ ew,
    const int* __restrict__ src, const int* __restrict__ dst,
    float* __restrict__ msg, long long total) {
    long long idx = (long long)blockIdx.x * 256 + threadIdx.x;
    if (idx >= total) return;
    const int e = (int)(idx >> 5);
    const int c = (int)(idx & 31);
    const float v = feat[(long long)src[e] * N_FIN + c] * ew[e];
    atomicAdd(&msg[(long long)dst[e] * N_FIN + c], v);
}

// ---------------------------------------------------------------------------
// Fused 3-layer MLP. 8 waves/block, 16 node-rows per wave (128 rows/block).
// Weights (f16) + biases (f32) LDS-resident; per-wave staging 11 KB.
// ---------------------------------------------------------------------------
__global__ __launch_bounds__(256) void gnn_fused(
    const float* __restrict__ feat, const float* __restrict__ msg,
    const float* __restrict__ Win, const float* __restrict__ bin,
    const float* __restrict__ Wh,  const float* __restrict__ bh,
    const float* __restrict__ Wo,  const float* __restrict__ bo,
    float* __restrict__ out, int nNodes) {
    __shared__ __align__(16) _Float16 sWin[HID * 2 * N_FIN];   // 256x64  (32 KB)
    __shared__ __align__(16) _Float16 sWh [HID * HID];         // 256x256 (128 KB)
    __shared__ __align__(16) _Float16 sWo [N_FOUT * HID];      // 64x256  (32 KB)
    __shared__ __align__(16) _Float16 sScr[8 * 5632];          // 8 waves (88 KB)
    __shared__ __align__(16) float    sBias[HID + HID + N_FOUT]; // 2.25 KB

    const int tid = threadIdx.x;
    for (int i = tid; i < HID * 2 * N_FIN; i += 256) sWin[i] = (_Float16)Win[i];
    for (int i = tid; i < HID * HID;       i += 256) sWh[i]  = (_Float16)Wh[i];
    for (int i = tid; i < N_FOUT * HID;    i += 256) sWo[i]  = (_Float16)Wo[i];
    if (tid < HID)    sBias[tid]           = bin[tid];
    if (tid < HID)    sBias[HID + tid]     = bh[tid];
    if (tid < N_FOUT) sBias[2 * HID + tid] = bo[tid];
    __syncthreads();

    const float* sBin = sBias;
    const float* sBh  = sBias + HID;
    const float* sBo  = sBias + 2 * HID;

    const int wave = tid >> 5;
    const int lane = tid & 31;
    _Float16* bufA = sScr + wave * 5632;   // 16x64  concat input tile
    _Float16* bufH = bufA + 16 * 64;       // 16x256 H1 (f16, post-ELU)
    _Float16* bufS = bufH + 16 * HID;      // 16x32  H2 K-chunk staging

    const int n0 = blockIdx.x * 128 + wave * 16;
    if (n0 >= nNodes) return;              // wave-uniform; EXEC stays all-1s

    // ---- build A0 = concat(feature, msg) as f16, row-major 16x64 ----------
#pragma unroll
    for (int it = 0; it < 32; ++it) {
        const int idx = it * 32 + lane;
        const int r = idx >> 6, c = idx & 63;
        const int node = n0 + r;
        float v = 0.0f;
        if (node < nNodes)
            v = (c < N_FIN) ? feat[(long long)node * N_FIN + c]
                            : msg[(long long)node * N_FIN + (c - N_FIN)];
        bufA[idx] = (_Float16)v;
    }

    // ---- layer 1: [16x64] x [64x256]^T, K=64 -> 2 WMMAs per out tile ------
    for (int t = 0; t < 16; ++t) {
        v8f acc = {0.f, 0.f, 0.f, 0.f, 0.f, 0.f, 0.f, 0.f};
        acc = WMMA(ld_frag(bufA, 64, 0),  ld_frag(sWin + t * 16 * 64, 64, 0),  acc);
        acc = WMMA(ld_frag(bufA, 64, 32), ld_frag(sWin + t * 16 * 64, 64, 32), acc);
        st_elu_f16(bufH + t * 16, HID, acc, sBin, t * 16);
    }

    // ---- layers 2+3 interleaved: H2 tiles consumed in pairs (K=32) --------
    v8f acc3[4];
#pragma unroll
    for (int t3 = 0; t3 < 4; ++t3)
        acc3[t3] = (v8f){0.f, 0.f, 0.f, 0.f, 0.f, 0.f, 0.f, 0.f};

    for (int t2 = 0; t2 < 16; ++t2) {
        v8f acc = {0.f, 0.f, 0.f, 0.f, 0.f, 0.f, 0.f, 0.f};
#pragma unroll
        for (int kk = 0; kk < 8; ++kk)
            acc = WMMA(ld_frag(bufH, HID, kk * 32),
                       ld_frag(sWh + t2 * 16 * HID, HID, kk * 32), acc);
        st_elu_f16(bufS + (t2 & 1) * 16, 32, acc, sBh, t2 * 16);
        if (t2 & 1) {                       // two H2 tiles ready -> K=32 step
            const v16h a3 = ld_frag(bufS, 32, 0);
            const int k0 = (t2 - 1) * 16;
#pragma unroll
            for (int t3 = 0; t3 < 4; ++t3)
                acc3[t3] = WMMA(a3, ld_frag(sWo + t3 * 16 * HID, HID, k0), acc3[t3]);
        }
    }

    // ---- output: bias add, f32 store --------------------------------------
#pragma unroll
    for (int t3 = 0; t3 < 4; ++t3) {
        const int col = t3 * 16 + (lane & 15);
        const int r0  = (lane >> 4) << 3;
        const float bv = sBo[col];
#pragma unroll
        for (int r = 0; r < 8; ++r) {
            const int node = n0 + r0 + r;
            if (node < nNodes)
                out[(long long)node * N_FOUT + col] = acc3[t3][r] + bv;
        }
    }
}

// ---------------------------------------------------------------------------
extern "C" void kernel_launch(void* const* d_in, const int* in_sizes, int n_in,
                              void* d_out, int out_size, void* d_ws, size_t ws_size,
                              hipStream_t stream) {
    const float* feat = (const float*)d_in[0];   // [N, 32]
    const float* ew   = (const float*)d_in[1];   // [E]
    const float* Win  = (const float*)d_in[2];   // [256, 64]
    const float* bin  = (const float*)d_in[3];   // [256]
    const float* Wh   = (const float*)d_in[4];   // [256, 256]
    const float* bh   = (const float*)d_in[5];   // [256]
    const float* Wo   = (const float*)d_in[6];   // [64, 256]
    const float* bo   = (const float*)d_in[7];   // [64]
    const int*   ei   = (const int*)d_in[8];     // [2, E]

    const int nNodes = in_sizes[0] / N_FIN;
    const int nEdges = in_sizes[1];
    float* msg = (float*)d_ws;                   // [N, 32] scratch

    hipMemsetAsync(msg, 0, (size_t)nNodes * N_FIN * sizeof(float), stream);

    const long long total = (long long)nEdges * 32;
    const int sblocks = (int)((total + 255) / 256);
    gnn_scatter<<<sblocks, 256, 0, stream>>>(feat, ew, ei, ei + nEdges, msg, total);

    const int gblocks = (nNodes + 127) / 128;
    gnn_fused<<<gblocks, 256, 0, stream>>>(feat, msg, Win, bin, Wh, bh, Wo, bo,
                                           (float*)d_out, nNodes);
}
